// TorchModel_26027501814360
// MI455X (gfx1250) — matrix operand order repaired and checked by reading, not compiled, not run on hardware
//
#include <hip/hip_runtime.h>

// ---------------------------------------------------------------------------
// Types for CDNA5 WMMA (wave32, 16x16x32 bf16 -> f32 accumulate)
// ---------------------------------------------------------------------------
typedef __bf16 bf16;
typedef __attribute__((ext_vector_type(8)))  bf16  bf16x8;
typedef __attribute__((ext_vector_type(16))) bf16  bf16x16;
typedef __attribute__((ext_vector_type(8)))  float floatx8;

// Load a 16-element bf16 fragment as two contiguous 8-element (16B) chunks.
__device__ __forceinline__ bf16x16 load_frag2(const bf16* __restrict__ base,
                                              int off0, int off1) {
  bf16x8 c0 = *(const bf16x8*)(base + off0);
  bf16x8 c1 = *(const bf16x8*)(base + off1);
  bf16x16 r;
#pragma unroll
  for (int i = 0; i < 8; ++i) { r[i] = c0[i]; r[8 + i] = c1[i]; }
  return r;
}

__device__ __forceinline__ floatx8 wmma_bf16(bf16x16 a, bf16x16 b, floatx8 c) {
  return __builtin_amdgcn_wmma_f32_16x16x32_bf16(false, a, false, b,
                                                 (short)0, c, false, false);
}

// ---------------------------------------------------------------------------
// fp32 -> bf16 conversion / zero-fill helpers
// ---------------------------------------------------------------------------
__global__ void f32_to_bf16_kernel(const float* __restrict__ in,
                                   bf16* __restrict__ out, long n) {
  long i = (long)blockIdx.x * blockDim.x + threadIdx.x;
  if (i < n) out[i] = (bf16)in[i];
}

__global__ void zero_bf16_kernel(bf16* __restrict__ p, int n) {
  int i = blockIdx.x * blockDim.x + threadIdx.x;
  if (i < n) p[i] = (bf16)0.0f;
}

// ---------------------------------------------------------------------------
// Generic GEMM:  C[M,N] = act( A[M,K] @ W[N,K]^T + bias[N] )
// Block tile 128(M) x 256(N); 8 waves (2 in M x 4 in N); each wave computes
// a 64x64 region = 4x4 WMMA tiles.  K-loop is ping-pong software pipelined
// (register double-buffered) so v_wmma overlaps in-flight global_load_b128.
// M,K multiples of 128/64 assumed; N bounds-guarded (final layer N=1000:
// out-of-range B rows are clamped to a valid row, results never stored).
// ---------------------------------------------------------------------------
template <typename OutT, bool RELU>
__global__ __launch_bounds__(256) void gemm_bf16_kernel(
    const bf16* __restrict__ A,     // [M,K] row-major
    const bf16* __restrict__ W,     // [N,K] row-major
    const float* __restrict__ bias, // [N]
    OutT* __restrict__ Cout,        // [M,ldc]
    int M, int N, int K, int ldc) {
  const int lane = threadIdx.x & 31;
  const int wid  = threadIdx.x >> 5;
  const int half = lane >> 4;   // 0 or 1
  const int l16  = lane & 15;

  const int  wm    = wid & 1;               // 2 waves in M
  const int  wn    = wid >> 1;              // 4 waves in N
  const long mbase = (long)blockIdx.y * 128 + wm * 64;
  const int  nbase = blockIdx.x * 256 + wn * 64;

  floatx8 acc[4][4];
#pragma unroll
  for (int i = 0; i < 4; ++i)
#pragma unroll
    for (int j = 0; j < 4; ++j)
#pragma unroll
      for (int r = 0; r < 8; ++r) acc[i][j][r] = 0.0f;

  const bf16* Arow[4];
  const bf16* Wrow[4];
#pragma unroll
  for (int i = 0; i < 4; ++i) {
    Arow[i] = A + (mbase + i * 16 + l16) * (long)K;
    int nj  = nbase + i * 16 + l16;
    if (nj >= N) nj = N - 1;                // clamp: safe, result not stored
    Wrow[i] = W + (long)nj * K;
  }

  const int aoff0 = 8 * half;        // A chunk offsets within a k-step
  const int aoff1 = 16 + 8 * half;
  const int boff  = 8 * half;        // B chunk offset within a k-step

  bf16x16 a0[4], b0[4], a1[4], b1[4];

#define LOAD_SET(aa, bb, kk)                                                \
  {                                                                         \
    _Pragma("unroll") for (int i = 0; i < 4; ++i) {                         \
      aa[i] = load_frag2(Arow[i], (kk) + aoff0, (kk) + aoff1);              \
      bb[i] = load_frag2(Wrow[i], (kk) + boff, (kk) + boff + 16);           \
    }                                                                       \
  }
#define WMMA_SET(aa, bb)                                                    \
  {                                                                         \
    _Pragma("unroll") for (int i = 0; i < 4; ++i)                           \
        _Pragma("unroll") for (int j = 0; j < 4; ++j)                       \
            acc[i][j] = wmma_bf16(aa[i], bb[j], acc[i][j]);                 \
  }

  LOAD_SET(a0, b0, 0);
  int k0 = 0;
#pragma unroll 1
  for (; k0 < K - 64; k0 += 64) {
    LOAD_SET(a1, b1, k0 + 32);   // prefetch while a0/b0 WMMAs issue
    WMMA_SET(a0, b0);
    LOAD_SET(a0, b0, k0 + 64);
    WMMA_SET(a1, b1);
  }
  LOAD_SET(a1, b1, K - 32);
  WMMA_SET(a0, b0);
  WMMA_SET(a1, b1);
#undef LOAD_SET
#undef WMMA_SET

  // Epilogue: C layout per lane: VGPR r -> m = r + 8*half, n = l16.
#pragma unroll
  for (int i = 0; i < 4; ++i) {
#pragma unroll
    for (int j = 0; j < 4; ++j) {
      const int n = nbase + j * 16 + l16;
      if (n < N) {
        const float bv = bias[n];
#pragma unroll
        for (int r = 0; r < 8; ++r) {
          const long m = mbase + i * 16 + r + 8 * half;
          float v = acc[i][j][r] + bv;
          if (RELU) v = fmaxf(v, 0.0f);
          Cout[m * (long)ldc + n] = (OutT)v;
        }
      }
    }
  }
}

// ---------------------------------------------------------------------------
// One GRU time step, fully fused (same ping-pong K pipelining):
//   gh = h_prev @ W_hh^T + b_hh  (3 gates, WMMA)
//   r = sig(xr+hr); z = sig(xz+hz); n = tanh(xn + r*hn)
//   h_next = (1-z)*n + z*h_prev ;  hs[:,t,:] = h_next
// Grid: H/64 blocks; 8 waves per block: (mt = w&1) 16 batch rows,
// (nt = w>>1) one 16-wide hidden-column subtile, 3 gate accumulators.
// ---------------------------------------------------------------------------
__global__ __launch_bounds__(256) void gru_step_kernel(
    const bf16* __restrict__ h_prev,  // [32,H] bf16
    bf16* __restrict__ h_next,        // [32,H] bf16
    const bf16* __restrict__ gx,      // [B*T, 3H] bf16 (includes b_ih)
    const bf16* __restrict__ Whh,     // [3H, H] bf16
    const float* __restrict__ b_hh,   // [3H]
    bf16* __restrict__ hs,            // [B*T, H] bf16
    int t) {
  const int H = 1024, T = 512;
  const int lane = threadIdx.x & 31;
  const int wid  = threadIdx.x >> 5;
  const int half = lane >> 4;
  const int l16  = lane & 15;

  const int mt    = wid & 1;                 // which 16 batch rows
  const int nt    = wid >> 1;                // 0..3 -> 16-col subtile
  const int jbase = blockIdx.x * 64 + nt * 16;
  const int j     = jbase + l16;

  const bf16* Arow = h_prev + (mt * 16 + l16) * H;
  const bf16* Wg[3];
#pragma unroll
  for (int g = 0; g < 3; ++g) Wg[g] = Whh + (long)(g * H + j) * H;

  floatx8 acc[3];
#pragma unroll
  for (int g = 0; g < 3; ++g)
#pragma unroll
    for (int r = 0; r < 8; ++r) acc[g][r] = 0.0f;

  const int aoff0 = 8 * half;
  const int aoff1 = 16 + 8 * half;
  const int boff  = 8 * half;

  bf16x16 a0, a1, g0[3], g1[3];

#define LOAD_SET(aa, gg, kk)                                                \
  {                                                                         \
    aa = load_frag2(Arow, (kk) + aoff0, (kk) + aoff1);                      \
    _Pragma("unroll") for (int g = 0; g < 3; ++g) gg[g] =                   \
        load_frag2(Wg[g], (kk) + boff, (kk) + boff + 16);                   \
  }
#define WMMA_SET(aa, gg)                                                    \
  {                                                                         \
    _Pragma("unroll") for (int g = 0; g < 3; ++g)                           \
        acc[g] = wmma_bf16(aa, gg[g], acc[g]);                              \
  }

  LOAD_SET(a0, g0, 0);
  int k0 = 0;
#pragma unroll 1
  for (; k0 < H - 64; k0 += 64) {
    LOAD_SET(a1, g1, k0 + 32);
    WMMA_SET(a0, g0);
    LOAD_SET(a0, g0, k0 + 64);
    WMMA_SET(a1, g1);
  }
  LOAD_SET(a1, g1, H - 32);
  WMMA_SET(a0, g0);
  WMMA_SET(a1, g1);
#undef LOAD_SET
#undef WMMA_SET

  const float bhr = b_hh[j];
  const float bhz = b_hh[H + j];
  const float bhn = b_hh[2 * H + j];

#pragma unroll
  for (int r = 0; r < 8; ++r) {
    const int  b    = mt * 16 + r + 8 * half;  // batch row 0..31
    const long grow = ((long)b * T + t) * (3L * H);
    const float xr = (float)gx[grow + j];
    const float xz = (float)gx[grow + H + j];
    const float xn = (float)gx[grow + 2 * H + j];
    const float hr = acc[0][r] + bhr;
    const float hz = acc[1][r] + bhz;
    const float hn = acc[2][r] + bhn;
    const float rg = 1.0f / (1.0f + __expf(-(xr + hr)));
    const float zg = 1.0f / (1.0f + __expf(-(xz + hz)));
    const float ng = tanhf(xn + rg * hn);
    const float ho = (float)h_prev[b * H + j];
    const float hv = (1.0f - zg) * ng + zg * ho;
    h_next[b * H + j] = (bf16)hv;
    hs[((long)b * T + t) * H + j] = (bf16)hv;
  }
}

// ---------------------------------------------------------------------------
// Host-side orchestration
// ---------------------------------------------------------------------------
extern "C" void kernel_launch(void* const* d_in, const int* in_sizes, int n_in,
                              void* d_out, int out_size, void* d_ws,
                              size_t ws_size, hipStream_t stream) {
  (void)in_sizes; (void)n_in; (void)out_size; (void)ws_size;
  const int B = 32, T = 512, D = 1024, H = 1024, C = 1000;
  const long M = (long)B * T;  // 16384

  const float* features = (const float*)d_in[0];
  const float* Wp   = (const float*)d_in[1];
  const float* bp   = (const float*)d_in[2];
  const float* W_ih = (const float*)d_in[3];
  const float* b_ih = (const float*)d_in[4];
  const float* W_hh = (const float*)d_in[5];
  const float* b_hh = (const float*)d_in[6];
  const float* W1   = (const float*)d_in[7];
  const float* b1   = (const float*)d_in[8];
  const float* W2   = (const float*)d_in[9];
  const float* b2   = (const float*)d_in[10];
  float* out = (float*)d_out;

  // Workspace carve-up (256B aligned). Total ~187 MB with aliasing.
  char* ws = (char*)d_ws;
  auto carve = [&](size_t bytes) -> char* {
    char* p = ws;
    ws += (bytes + 255) & ~(size_t)255;
    return p;
  };
  bf16* feat_bf = (bf16*)carve((size_t)M * D * 2);     // later reused as out1
  bf16* Wp_bf   = (bf16*)carve((size_t)H * D * 2);
  bf16* Wih_bf  = (bf16*)carve((size_t)3 * H * H * 2);
  bf16* Whh_bf  = (bf16*)carve((size_t)3 * H * H * 2);
  bf16* W1_bf   = (bf16*)carve((size_t)H * H * 2);
  bf16* W2_bf   = (bf16*)carve((size_t)C * H * 2);
  bf16* x_bf    = (bf16*)carve((size_t)M * H * 2);     // later reused as hs
  bf16* gx_bf   = (bf16*)carve((size_t)M * 3 * H * 2);
  bf16* h0      = (bf16*)carve((size_t)B * H * 2);
  bf16* h1      = (bf16*)carve((size_t)B * H * 2);

  auto cvt = [&](const float* in, bf16* o, long n) {
    unsigned blocks = (unsigned)((n + 255) / 256);
    f32_to_bf16_kernel<<<dim3(blocks), dim3(256), 0, stream>>>(in, o, n);
  };
  cvt(features, feat_bf, M * D);
  cvt(Wp, Wp_bf, (long)H * D);
  cvt(W_ih, Wih_bf, 3L * H * H);
  cvt(W_hh, Whh_bf, 3L * H * H);
  cvt(W1, W1_bf, (long)H * H);
  cvt(W2, W2_bf, (long)C * H);
  zero_bf16_kernel<<<dim3((B * H + 255) / 256), dim3(256), 0, stream>>>(
      h0, B * H);

  const dim3 blk(256);
  // x = relu(feat @ Wp^T + bp)
  gemm_bf16_kernel<bf16, true><<<dim3(H / 256, M / 128), blk, 0, stream>>>(
      feat_bf, Wp_bf, bp, x_bf, (int)M, H, D, H);
  // gx = x @ W_ih^T + b_ih
  gemm_bf16_kernel<bf16, false><<<dim3(3 * H / 256, M / 128), blk, 0, stream>>>(
      x_bf, Wih_bf, b_ih, gx_bf, (int)M, 3 * H, H, 3 * H);

  // GRU scan: one fused kernel per timestep; double-buffered hidden state.
  bf16* hs_bf = x_bf;  // x is dead after the gx GEMM
  bf16* hp = h0;
  bf16* hn = h1;
  for (int t = 0; t < T; ++t) {
    gru_step_kernel<<<dim3(H / 64), blk, 0, stream>>>(hp, hn, gx_bf, Whh_bf,
                                                      b_hh, hs_bf, t);
    bf16* tmp = hp; hp = hn; hn = tmp;
  }

  // out1 = relu(hs @ W1^T + b1)   (reuse feat buffer)
  bf16* out1_bf = feat_bf;
  gemm_bf16_kernel<bf16, true><<<dim3(H / 256, M / 128), blk, 0, stream>>>(
      hs_bf, W1_bf, b1, out1_bf, (int)M, H, H, H);
  // out = out1 @ W2^T + b2   (fp32 output, N=1000 guarded)
  gemm_bf16_kernel<float, false>
      <<<dim3((C + 255) / 256, M / 128), blk, 0, stream>>>(
          out1_bf, W2_bf, b2, out, (int)M, C, H, C);
}
